// SpikingTransformer_11416023073065
// MI455X (gfx1250) — matrix operand order repaired and checked by reading, not compiled, not run on hardware
//
#include <hip/hip_runtime.h>
#include <math.h>

// ---------------------------------------------------------------- types
typedef __attribute__((ext_vector_type(16))) __bf16 v16bf;
typedef __attribute__((ext_vector_type(8)))  float  v8f;
typedef unsigned __attribute__((ext_vector_type(4))) v4u;
typedef int      __attribute__((ext_vector_type(8))) v8i_t;
typedef int      __attribute__((ext_vector_type(4))) v4i_t;

struct alignas(8)  US4 { unsigned short v[4]; };
struct alignas(16) F4  { float v[4]; };

__device__ __forceinline__ unsigned short f2bf(float f) {
  union { float f; unsigned u; } c; c.f = f;
  unsigned r = c.u + 0x7FFFu + ((c.u >> 16) & 1u);   // round-to-nearest-even
  return (unsigned short)(r >> 16);
}
__device__ __forceinline__ float bf2f(unsigned short h) {
  union { unsigned u; float f; } c; c.u = ((unsigned)h) << 16; return c.f;
}
__device__ __forceinline__ __bf16 us2bf(unsigned short u) {
  union { unsigned short u; __bf16 b; } c; c.u = u; return c.b;
}

// CDNA5 async global->LDS copy (ASYNCcnt-tracked, bypasses VGPRs).
__device__ __forceinline__ void async_copy_b128(unsigned lds_off, const void* gptr) {
  unsigned long long ga = (unsigned long long)gptr;
  asm volatile("global_load_async_to_lds_b128 %0, %1, off"
               :: "v"(lds_off), "v"(ga) : "memory");
}
// Async loads complete in order: waiting <=2 retires the previous tile's A ops.
__device__ __forceinline__ void wait_async(bool pipelined) {
  if (pipelined) asm volatile("s_wait_asynccnt 0x2" ::: "memory");
  else           asm volatile("s_wait_asynccnt 0x0" ::: "memory");
}

// CDNA5 Tensor Data Mover: 2D tile (tile_d1 rows x tile_d0 elems, bf16) from
// global (row stride = stride0 elements) into LDS at ldsOff, inserting
// pad_amount DWORDs of padding every pad_interval DWORDs (ISA D# 8.3/8.4).
// Tensor dims set huge => no OOB clipping (tiles never exceed real tensors).
__device__ __forceinline__ void tdm_load_2d(unsigned ldsOff, const void* g,
    unsigned tile_d0, unsigned tile_d1, unsigned long long stride0,
    unsigned padIv, unsigned padAmt)
{
  unsigned long long ga = (unsigned long long)g;
  v4u g0;
  g0[0] = 1u;                                    // count=1, user D#
  g0[1] = ldsOff;                                // lds_addr (bytes)
  g0[2] = (unsigned)ga;                          // global_addr[31:0]
  g0[3] = (unsigned)(ga >> 32) | (2u << 30);     // global_addr[56:32] | type=2
  v8i_t g1;
  g1[0] = (int)((1u << 16)                       // data_size = 2 bytes
              | (1u << 20)                       // pad_enable
              | (padIv  << 22)                   // pad_interval
              | (padAmt << 25));                 // pad_amount
  g1[1] = (int)0xFFFF0000u;                      // tensor_dim0 lo16 << 16
  g1[2] = (int)(0x7FFFu | 0xFFFF0000u);          // dim0 hi16 | dim1 lo16<<16
  g1[3] = (int)(0x7FFFu | (tile_d0 << 16));      // dim1 hi16 | tile_dim0
  g1[4] = (int)tile_d1;                          // tile_dim1 (tile_dim2=0)
  g1[5] = (int)(unsigned)stride0;                // tensor_dim0_stride[31:0]
  g1[6] = (int)((unsigned)(stride0 >> 32) & 0xFFFFu); // stride[47:32]
  g1[7] = 0;
  v4i_t z4 = {0, 0, 0, 0};
  v8i_t z8 = {0, 0, 0, 0, 0, 0, 0, 0};
  __builtin_amdgcn_tensor_load_to_lds(g0, g1, z4, z4, z8, 0);
}

// ---------------------------------------------------------------- GEMM
// C[M,N] = A[M,K](bf16) x B[K,N](bf16), double-buffered:
//   A staged by per-thread global_load_async_to_lds_b128 (ASYNCcnt),
//   B staged by one TDM tensor_load_to_lds per k-step (TENSORcnt, wave 0).
// TRB==0: B row-major [K,N];  TRB==1: B(k,n)=Bsrc[n*ldb+k]  (Q*K^T).
// EPI: 0 bias->Cf | 1 bias->Cbf | 2 sigmoid(scale*acc)->Cbf | 3 ->Cbf
//    | 4 bias+residual->Cf
// Workgroup tile 128x64, 8 waves (4x2), each wave 2x2 of 16x16x32 WMMAs.
constexpr int AS_TILE = 128 * 48;   // row stride 48 (96B, 16B-aligned)
constexpr int BS_TILE = 64 * 48;    // TRB0 uses [32][72]; TRB1 uses [64][48]

__device__ __forceinline__ void stage_A(
    const unsigned short* __restrict__ A, int lda, int tM,
    int k0, unsigned aB, int tid)
{
#pragma unroll
  for (int i = 0; i < 2; ++i) {            // A: 128x32, 2 x b128 per thread
    int idx = tid + i * 256;
    int r = idx >> 2, c = (idx & 3) << 3;
    async_copy_b128(aB + (unsigned)(r * 48 + c) * 2,
                    A + (size_t)(tM + r) * lda + (k0 + c));
  }
}

template<int TRB>
__device__ __forceinline__ void stage_B_tdm(
    const unsigned short* __restrict__ B, int ldb, int tN,
    int k0, unsigned bB)
{
  if (TRB == 0)   // 32 rows x 64 elems (128B=32dw/row), LDS stride 144B: pad 4dw/32dw
    tdm_load_2d(bB, B + (size_t)k0 * ldb + tN, 64, 32, (unsigned long long)ldb, 4, 3);
  else            // 64 rows x 32 elems (64B=16dw/row),  LDS stride  96B: pad 8dw/16dw
    tdm_load_2d(bB, B + (size_t)tN * ldb + k0, 32, 64, (unsigned long long)ldb, 3, 7);
}

template<int TRB, int EPI>
__global__ __launch_bounds__(256) void gemm_bf16_wmma(
    const unsigned short* __restrict__ A, int lda, long long sAb, long long sAh,
    const unsigned short* __restrict__ B, int ldb, long long sBb, long long sBh,
    const float* __restrict__ bias,
    const float* __restrict__ residual,
    float* __restrict__ Cf, unsigned short* __restrict__ Cbf,
    int ldc, long long sCb, long long sCh,
    int M, int N, int K, float scale, int Hdiv)
{
  __shared__ __attribute__((aligned(16))) unsigned short As[2 * AS_TILE];
  __shared__ __attribute__((aligned(16))) unsigned short Bsh[2 * BS_TILE];

  const int tid = threadIdx.x;
  const int z  = blockIdx.z;
  const int zb = z / Hdiv, zh = z % Hdiv;
  A += zb * sAb + zh * sAh;
  B += zb * sBb + zh * sBh;
  const long long cOff = (long long)zb * sCb + (long long)zh * sCh;
  if (EPI == 0 || EPI == 4) Cf += cOff; else Cbf += cOff;
  if (EPI == 4) residual += cOff;

  const int tM = blockIdx.y * 128;
  const int tN = blockIdx.x * 64;

  const int wave = tid >> 5, lane = tid & 31;
  const int half = lane >> 4, l16 = lane & 15;
  const int wm = wave >> 1, wn = wave & 1;   // 4x2 wave grid

  const unsigned aB0 = (unsigned)(unsigned long long)(const void*)As;
  const unsigned bB0 = (unsigned)(unsigned long long)(const void*)Bsh;

  v8f acc00 = {}, acc01 = {}, acc10 = {}, acc11 = {};

  stage_A(A, lda, tM, 0, aB0, tid);
  if (tid < 32) stage_B_tdm<TRB>(B, ldb, tN, 0, bB0);

  int cur = 0;
  for (int k0 = 0; k0 < K; k0 += 32, cur ^= 1) {
    const bool more = (k0 + 32) < K;
    if (more) {
      stage_A(A, lda, tM, k0 + 32, aB0 + (unsigned)(cur ^ 1) * AS_TILE * 2, tid);
      if (tid < 32)
        stage_B_tdm<TRB>(B, ldb, tN, k0 + 32, bB0 + (unsigned)(cur ^ 1) * BS_TILE * 2);
      __builtin_amdgcn_s_wait_tensorcnt(1);   // previous tile's TDM retired
    } else {
      __builtin_amdgcn_s_wait_tensorcnt(0);
    }
    wait_async(more);
    __syncthreads();

    const unsigned short* Ab = As  + cur * AS_TILE;
    const unsigned short* Bb = Bsh + cur * BS_TILE;

    // ---- wave32 fragments: k = 16*half + j for both A and B (contiguous order)
    v16bf a0, a1, b0, b1;
#pragma unroll
    for (int j = 0; j < 16; ++j) {
      int ka = j + (half << 4);                           // A: K index
      a0[j] = us2bf(Ab[(wm * 32 + l16) * 48 + ka]);
      a1[j] = us2bf(Ab[(wm * 32 + 16 + l16) * 48 + ka]);
      int kb = j + (half << 4);                           // B: K index
      if (TRB == 0) {
        b0[j] = us2bf(Bb[kb * 72 + wn * 32 + l16]);
        b1[j] = us2bf(Bb[kb * 72 + wn * 32 + 16 + l16]);
      } else {
        b0[j] = us2bf(Bb[(wn * 32 + l16) * 48 + kb]);
        b1[j] = us2bf(Bb[(wn * 32 + 16 + l16) * 48 + kb]);
      }
    }
    acc00 = __builtin_amdgcn_wmma_f32_16x16x32_bf16(false, a0, false, b0, (short)0, acc00, false, false);
    acc01 = __builtin_amdgcn_wmma_f32_16x16x32_bf16(false, a0, false, b1, (short)0, acc01, false, false);
    acc10 = __builtin_amdgcn_wmma_f32_16x16x32_bf16(false, a1, false, b0, (short)0, acc10, false, false);
    acc11 = __builtin_amdgcn_wmma_f32_16x16x32_bf16(false, a1, false, b1, (short)0, acc11, false, false);
    __syncthreads();   // waves done reading before next iteration overwrites
  }

  // ---- epilogue: C layout VGPR r -> M = r + 8*half, N = lane&15
  v8f accs[2][2] = { { acc00, acc01 }, { acc10, acc11 } };
#pragma unroll
  for (int i = 0; i < 2; ++i)
#pragma unroll
    for (int j = 0; j < 2; ++j)
#pragma unroll
      for (int r = 0; r < 8; ++r) {
        int mg = tM + wm * 32 + i * 16 + r + 8 * half;
        int ng = tN + wn * 32 + j * 16 + l16;
        float val = accs[i][j][r];
        if (EPI == 0 || EPI == 1 || EPI == 4) val += bias[ng];
        if (EPI == 2) val = __builtin_amdgcn_rcpf(1.0f + __expf(-val * scale));
        if (EPI == 4) val += residual[(size_t)mg * ldc + ng];
        size_t o = (size_t)mg * ldc + ng;
        if (EPI == 0 || EPI == 4) Cf[o] = val;
        else                      Cbf[o] = f2bf(val);
      }
}

// ---------------------------------------------------------------- fp32 -> bf16
__global__ void cvt_bf16_kernel(const float* __restrict__ in,
                                unsigned short* __restrict__ out, size_t n)
{
  size_t i = ((size_t)blockIdx.x * blockDim.x + threadIdx.x) * 4;
  if (i + 3 < n) {
    F4 v = *(const F4*)(in + i);
    US4 o;
    o.v[0] = f2bf(v.v[0]); o.v[1] = f2bf(v.v[1]);
    o.v[2] = f2bf(v.v[2]); o.v[3] = f2bf(v.v[3]);
    *(US4*)(out + i) = o;
  } else {
    for (; i < n; ++i) out[i] = f2bf(in[i]);
  }
}

// ---------------------------------------------------------------- layernorm
__global__ __launch_bounds__(256) void layernorm_kernel(
    const float* __restrict__ inF, const unsigned short* __restrict__ inB,
    const float* __restrict__ g, const float* __restrict__ b,
    unsigned short* __restrict__ out, int D)
{
  __shared__ float s1[256], s2[256];
  const int row = blockIdx.x, tid = threadIdx.x;
  float sum = 0.f, sq = 0.f;
  for (int c = tid; c < D; c += 256) {
    float v = inF ? inF[(size_t)row * D + c] : bf2f(inB[(size_t)row * D + c]);
    sum += v; sq += v * v;
  }
  s1[tid] = sum; s2[tid] = sq; __syncthreads();
  for (int s = 128; s > 0; s >>= 1) {
    if (tid < s) { s1[tid] += s1[tid + s]; s2[tid] += s2[tid + s]; }
    __syncthreads();
  }
  const float mu = s1[0] / (float)D;
  const float var = s2[0] / (float)D - mu * mu;
  const float rstd = rsqrtf(var + 1e-5f);
  for (int c = tid; c < D; c += 256) {
    float v = inF ? inF[(size_t)row * D + c] : bf2f(inB[(size_t)row * D + c]);
    out[(size_t)row * D + c] = f2bf((v - mu) * rstd * g[c] + b[c]);
  }
}

// ---------------------------------------------------------------- LIF (pointwise)
__global__ void lif_point_kernel(const float* __restrict__ pre,
    float* __restrict__ mem, float* __restrict__ bb,
    unsigned short* __restrict__ sOut, size_t n)
{
  size_t i = (size_t)blockIdx.x * blockDim.x + threadIdx.x;
  if (i >= n) return;
  float m = 0.9f * mem[i] + pre[i];
  float thr = 1.0f + 1.8f * bb[i];
  float s = (m - thr >= 0.0f) ? 1.0f : 0.0f;
  m -= s * thr;
  bb[i]  = 0.9f * bb[i] + 0.1f * s;
  mem[i] = m;
  sOut[i] = f2bf(s);
}

// ---------------------------------------------------------------- LIF (token scan)
__global__ void lif_scan_kernel(const float* __restrict__ x,
    float* __restrict__ mem, float* __restrict__ bb,
    unsigned short* __restrict__ outB, float* __restrict__ outF,
    int Bn, int S, int F)
{
  int j = blockIdx.x * blockDim.x + threadIdx.x;
  if (j >= Bn * F) return;
  int b = j / F, f = j % F;
  float m = mem[j], a = bb[j];
  for (int t = 0; t < S; ++t) {
    size_t idx = ((size_t)(b * S + t)) * F + f;
    m = 0.9f * m + x[idx];
    float thr = 1.0f + 1.8f * a;
    float s = (m - thr >= 0.0f) ? 1.0f : 0.0f;
    m -= s * thr;
    a = 0.9f * a + 0.1f * s;
    if (outB) outB[idx] = f2bf(s);
    if (outF) outF[idx] = s;
  }
  mem[j] = m; bb[j] = a;
}

// ---------------------------------------------------------------- embedding
__global__ void embed_kernel(const int* __restrict__ ids,
    const float* __restrict__ tok, const float* __restrict__ pos,
    float* __restrict__ x, int S, int D)
{
  int row = blockIdx.x;
  int s = row % S;
  int id = ids[row];
  for (int c = threadIdx.x; c < D; c += blockDim.x)
    x[(size_t)row * D + c] = tok[(size_t)id * D + c] + pos[(size_t)s * D + c];
}

// ---------------------------------------------------------------- host side
static inline void launch_gemm(hipStream_t st, int trB, int epi,
  const unsigned short* A, int lda, long long sAb, long long sAh,
  const unsigned short* B, int ldb, long long sBb, long long sBh,
  const float* bias, const float* res,
  float* Cf, unsigned short* Cbf, int ldc, long long sCb, long long sCh,
  int M, int N, int K, float scale, int batch, int Hdiv)
{
  dim3 g(N / 64, M / 128, batch);
#define GEMM_GO(T, E) gemm_bf16_wmma<T, E><<<g, 256, 0, st>>>( \
    A, lda, sAb, sAh, B, ldb, sBb, sBh, bias, res, Cf, Cbf, ldc, sCb, sCh, \
    M, N, K, scale, Hdiv)
  if (trB) { GEMM_GO(1, 2); }
  else switch (epi) {
    case 0:  GEMM_GO(0, 0); break;
    case 1:  GEMM_GO(0, 1); break;
    case 3:  GEMM_GO(0, 3); break;
    default: GEMM_GO(0, 4); break;
  }
#undef GEMM_GO
}

static inline void launch_cvt(hipStream_t st, const float* in, unsigned short* out, size_t n)
{
  size_t threads = (n + 3) / 4;
  cvt_bf16_kernel<<<(unsigned)((threads + 255) / 256), 256, 0, st>>>(in, out, n);
}

extern "C" void kernel_launch(void* const* d_in, const int* in_sizes, int n_in,
                              void* d_out, int out_size, void* d_ws, size_t ws_size,
                              hipStream_t stream)
{
  (void)in_sizes; (void)n_in; (void)out_size; (void)ws_size;
  constexpr int Bn = 4, S = 512, D = 512, H = 8, dh = 64, FF = 2048, L = 4, V = 32000;
  constexpr int M  = Bn * S;                 // 2048 token rows
  constexpr size_t MD  = (size_t)M * D;
  constexpr size_t MFF = (size_t)M * FF;

  const int*   ids    = (const int*)  d_in[0];
  const float* tok    = (const float*)d_in[1];
  const float* pos    = (const float*)d_in[2];
  const float* ln1_g  = (const float*)d_in[3];
  const float* ln1_b  = (const float*)d_in[4];
  const float* Wq     = (const float*)d_in[5];
  const float* bq     = (const float*)d_in[6];
  const float* Wk     = (const float*)d_in[7];
  const float* bk     = (const float*)d_in[8];
  const float* Wv     = (const float*)d_in[9];
  const float* bv     = (const float*)d_in[10];
  const float* Wo     = (const float*)d_in[11];
  const float* bo     = (const float*)d_in[12];
  const float* ln2_g  = (const float*)d_in[13];
  const float* ln2_b  = (const float*)d_in[14];
  const float* W1     = (const float*)d_in[15];
  const float* b1     = (const float*)d_in[16];
  const float* W2     = (const float*)d_in[17];
  const float* b2     = (const float*)d_in[18];
  const float* lnf_g  = (const float*)d_in[19];
  const float* lnf_b  = (const float*)d_in[20];
  const float* Wout   = (const float*)d_in[21];
  const float* bout   = (const float*)d_in[22];
  float* out          = (float*)d_out;

  // ---- workspace allocator
  char* wp = (char*)d_ws;
  auto alloc = [&](size_t bytes) -> char* {
    char* p = wp;
    wp += (bytes + 255) & ~(size_t)255;
    return p;
  };
  float*          x     = (float*)         alloc(MD * 4);       // master activations
  unsigned short* h     = (unsigned short*)alloc(MD * 2);       // LN output (bf16)
  float*          pre   = (float*)         alloc(MFF * 4);      // reused fp32 scratch
  unsigned short* sq    = (unsigned short*)alloc(MD * 2);
  unsigned short* sk    = (unsigned short*)alloc(MD * 2);
  unsigned short* vbf   = (unsigned short*)alloc(MD * 2);
  unsigned short* ao    = (unsigned short*)alloc(MD * 2);
  unsigned short* attn  = (unsigned short*)alloc((size_t)Bn * H * S * S * 2);
  unsigned short* sp1b  = (unsigned short*)alloc(MD * 2);
  float*          sp1f  = (float*)         alloc(MD * 4);
  unsigned short* sp2b  = (unsigned short*)alloc(MFF * 2);
  float*          stQK  = (float*)         alloc((size_t)L * 4 * MD * 4);   // q/k mem+b
  float*          stSc  = (float*)         alloc((size_t)L * 24576 * 4);    // scan states
  // bf16 weight copies (pure-copy TDM staging in GEMM, halves weight bytes)
  unsigned short* wqB   = (unsigned short*)alloc((size_t)L * D * D  * 2);
  unsigned short* wkB   = (unsigned short*)alloc((size_t)L * D * D  * 2);
  unsigned short* wvB   = (unsigned short*)alloc((size_t)L * D * D  * 2);
  unsigned short* woB   = (unsigned short*)alloc((size_t)L * D * D  * 2);
  unsigned short* w1B   = (unsigned short*)alloc((size_t)L * D * FF * 2);
  unsigned short* w2B   = (unsigned short*)alloc((size_t)L * FF * D * 2);
  unsigned short* woutB = (unsigned short*)alloc((size_t)D * V * 2);

  (void)hipMemsetAsync(stQK, 0, (size_t)L * 4 * MD * 4, stream);
  (void)hipMemsetAsync(stSc, 0, (size_t)L * 24576 * 4, stream);

  launch_cvt(stream, Wq,   wqB,   (size_t)L * D * D);
  launch_cvt(stream, Wk,   wkB,   (size_t)L * D * D);
  launch_cvt(stream, Wv,   wvB,   (size_t)L * D * D);
  launch_cvt(stream, Wo,   woB,   (size_t)L * D * D);
  launch_cvt(stream, W1,   w1B,   (size_t)L * D * FF);
  launch_cvt(stream, W2,   w2B,   (size_t)L * FF * D);
  launch_cvt(stream, Wout, woutB, (size_t)D * V);

  // ---- embedding
  embed_kernel<<<M, 256, 0, stream>>>(ids, tok, pos, x, S, D);

  const float scaleAttn = 1.0f / 8.0f;     // 1/sqrt(dh)
  const long long SD = (long long)S * D, HSS = (long long)H * S * S, SS = (long long)S * S;

  for (int t = 0; t < 2; ++t) {
    for (int l = 0; l < L; ++l) {
      float* qm = stQK + ((size_t)l * 4 + 0) * MD;
      float* qb = stQK + ((size_t)l * 4 + 1) * MD;
      float* km = stQK + ((size_t)l * 4 + 2) * MD;
      float* kb2= stQK + ((size_t)l * 4 + 3) * MD;
      float* sc = stSc + (size_t)l * 24576;
      float *l1m = sc, *l1b = sc + 2048, *l2m = sc + 4096, *l2b = sc + 12288,
            *l3m = sc + 20480, *l3b = sc + 22528;

      // h = LN1(x)
      layernorm_kernel<<<M, 256, 0, stream>>>(x, nullptr, ln1_g + l * D, ln1_b + l * D, h, D);

      // q_pre = h @ Wq + bq ; LIF pointwise -> sq
      launch_gemm(stream, 0, 0, h, D, 0, 0, wqB + (size_t)l * D * D, D, 0, 0,
                  bq + l * D, nullptr, pre, nullptr, D, 0, 0, M, D, D, 1.f, 1, 1);
      lif_point_kernel<<<(MD + 255) / 256, 256, 0, stream>>>(pre, qm, qb, sq, MD);

      // k_pre = h @ Wk + bk ; LIF pointwise -> sk
      launch_gemm(stream, 0, 0, h, D, 0, 0, wkB + (size_t)l * D * D, D, 0, 0,
                  bk + l * D, nullptr, pre, nullptr, D, 0, 0, M, D, D, 1.f, 1, 1);
      lif_point_kernel<<<(MD + 255) / 256, 256, 0, stream>>>(pre, km, kb2, sk, MD);

      // v = h @ Wv + bv  (bf16 out)
      launch_gemm(stream, 0, 1, h, D, 0, 0, wvB + (size_t)l * D * D, D, 0, 0,
                  bv + l * D, nullptr, nullptr, vbf, D, 0, 0, M, D, D, 1.f, 1, 1);

      // attn = sigmoid(scale * sq @ sk^T)   [batched over (b,h), B transposed]
      launch_gemm(stream, 1, 2, sq, D, SD, dh, sk, D, SD, dh,
                  nullptr, nullptr, nullptr, attn, S, HSS, SS,
                  S, S, dh, scaleAttn, Bn * H, H);

      // ao = attn @ v   [batched over (b,h)]
      launch_gemm(stream, 0, 3, attn, S, HSS, SS, vbf, D, SD, dh,
                  nullptr, nullptr, nullptr, ao, D, SD, dh,
                  S, dh, S, 1.f, Bn * H, H);

      // xa = x + ao @ Wo + bo   (into pre)
      launch_gemm(stream, 0, 4, ao, D, 0, 0, woB + (size_t)l * D * D, D, 0, 0,
                  bo + l * D, x, pre, nullptr, D, 0, 0, M, D, D, 1.f, 1, 1);

      // sp1 = lif_scan(xa)
      lif_scan_kernel<<<(Bn * D + 255) / 256, 256, 0, stream>>>(
          pre, l1m, l1b, sp1b, sp1f, Bn, S, D);

      // h = LN2(sp1)
      layernorm_kernel<<<M, 256, 0, stream>>>(nullptr, sp1b, ln2_g + l * D, ln2_b + l * D, h, D);

      // h1 = h @ W1 + b1   (into pre, M x FF)
      launch_gemm(stream, 0, 0, h, D, 0, 0, w1B + (size_t)l * D * FF, FF, 0, 0,
                  b1 + l * FF, nullptr, pre, nullptr, FF, 0, 0, M, FF, D, 1.f, 1, 1);

      // sp2 = lif_scan(h1)
      lif_scan_kernel<<<(Bn * FF + 255) / 256, 256, 0, stream>>>(
          pre, l2m, l2b, sp2b, nullptr, Bn, S, FF);

      // xf = sp1 + sp2 @ W2 + b2   (into pre[0:MD])
      launch_gemm(stream, 0, 4, sp2b, FF, 0, 0, w2B + (size_t)l * FF * D, D, 0, 0,
                  b2 + l * D, sp1f, pre, nullptr, D, 0, 0, M, D, FF, 1.f, 1, 1);

      // x = lif_scan(xf)
      lif_scan_kernel<<<(Bn * D + 255) / 256, 256, 0, stream>>>(
          pre, l3m, l3b, nullptr, x, Bn, S, D);
    }
  }

  // logits = LN_f(x) @ Wout + bout
  layernorm_kernel<<<M, 256, 0, stream>>>(x, nullptr, lnf_g, lnf_b, h, D);
  launch_gemm(stream, 0, 0, h, D, 0, 0, woutB, V, 0, 0,
              bout, nullptr, out, nullptr, V, 0, 0, M, V, D, 1.f, 1, 1);
}
